// DGAModel_44976897524612
// MI455X (gfx1250) — compile-verified, matrix operand order checked by reading, not run on hardware
//
#include <hip/hip_runtime.h>
#include <math.h>
#include <stdint.h>

// ---------------- problem constants ----------------
#define Bsz   64
#define Ssz   256
#define Dsz   512
#define Hh    8
#define DKk   64
#define FFs   2048
#define Ll    2
#define Mrows (Bsz * Ssz)        // 16384 token rows

typedef _Float16 half_t;
typedef __attribute__((ext_vector_type(16))) _Float16 v16h;
typedef __attribute__((ext_vector_type(8)))  float    v8f;

#define LDF 40   // LDS row stride in halfs: 32 payload + 8 pad (16B-aligned rows)

// output modes for the generic GEMM
#define MODE_F32      0
#define MODE_F16      1
#define MODE_F16_GELU 2
#define MODE_VT       3   // f16, scattered per-head transposed: vt[bh][dv][s]

union FragU { v16h v; unsigned int u[8]; };

// Load a 16x32 f16 WMMA A/B fragment from LDS laid out row-major, stride LDF.
// ISA layout: lane L holds row M = L%16; K-halves selected by L/16 -> the 8
// b32 loads merge into ds_load_b128 pairs.
__device__ inline v16h load_frag(const half_t* __restrict__ base, int lane) {
  FragU f;
  const half_t* row = base + (lane & 15) * LDF;
  const int hsel = (lane >> 4) << 3;
#pragma unroll
  for (int v = 0; v < 8; ++v) {
    int k = ((v >> 2) << 4) + hsel + ((v & 3) << 1);
    f.u[v] = *(const unsigned int*)(row + k);
  }
  return f.v;
}

#define WMMA(A_, B_, C_) \
  __builtin_amdgcn_wmma_f32_16x16x32_f16(false, (A_), false, (B_), (short)0, (C_), false, false)

__device__ inline float gelu_f(float x) {
  return 0.5f * x * (1.0f + erff(x * 0.70710678118654752440f));
}

// ---------------- CDNA5 async global->LDS staging ----------------
// Generic-pointer low 32 bits == LDS byte address.
__device__ inline uint32_t lds_off(const void* p) {
  return (uint32_t)(uintptr_t)p;
}
// one 16-byte async copy per lane; tracked by ASYNCcnt
__device__ inline void async_b128(uint32_t lds, const half_t* g) {
  asm volatile("global_load_async_to_lds_b128 %0, %1, off"
               :: "v"(lds), "v"(g) : "memory");
}
__device__ inline void wait_async() {
  asm volatile("s_wait_asynccnt 0x0" ::: "memory");
}

// stage 128 rows x 32 halfs (row stride lda halfs): 2 x b128 per thread
__device__ inline void stage_a_async(half_t* __restrict__ dst, const half_t* __restrict__ src,
                                     int lda, int tid) {
  __builtin_prefetch(src + 64, 0, 0);   // speculative: tile s+2
#pragma unroll
  for (int j = 0; j < 2; ++j) {
    const int chunk = tid * 2 + j;
    const int r = chunk >> 2;
    const int c = (chunk & 3) << 3;
    async_b128(lds_off(dst + r * LDF + c), src + (size_t)r * lda + c);
  }
}
// stage 64 rows x 32 halfs: 1 x b128 per thread
__device__ inline void stage_b_async(half_t* __restrict__ dst, const half_t* __restrict__ src,
                                     int ldb, int tid) {
  const int r = tid >> 2;
  const int c = (tid & 3) << 3;
  async_b128(lds_off(dst + r * LDF + c), src + (size_t)r * ldb + c);
}

// Per-wave 32x32 compute step: 2 A frags + 2 B frags -> 4 WMMAs (A reused)
__device__ inline void wave_mma(const half_t* __restrict__ As, const half_t* __restrict__ Bs,
                                int rowPair, int colHalf, int lane,
                                v8f& a00, v8f& a01, v8f& a10, v8f& a11) {
  v16h fa0 = load_frag(As + (rowPair * 32) * LDF, lane);
  v16h fa1 = load_frag(As + (rowPair * 32 + 16) * LDF, lane);
  v16h fb0 = load_frag(Bs + (colHalf * 32) * LDF, lane);
  v16h fb1 = load_frag(Bs + (colHalf * 32 + 16) * LDF, lane);
  a00 = WMMA(fa0, fb0, a00);
  a01 = WMMA(fa0, fb1, a01);
  a10 = WMMA(fa1, fb0, a10);
  a11 = WMMA(fa1, fb1, a11);
}

// =====================================================================
// Generic f16 GEMM: C = act(A[M][K] @ Bt[N][K]^T + bias[N]).
// Block tile 128x64, wave tile 32x32, async double-buffered LDS.
// grid = (N/64, M/128), block = 256.
// =====================================================================
__global__ __launch_bounds__(256) void gemm_f16(
    const half_t* __restrict__ A, const half_t* __restrict__ Bt,
    const float* __restrict__ bias, void* __restrict__ Cout,
    int M, int N, int K, int mode) {
  __shared__ __align__(16) half_t As[2][128 * LDF];
  __shared__ __align__(16) half_t Bs[2][64 * LDF];

  const int tid  = threadIdx.x;
  const int lane = tid & 31;
  const int wave = tid >> 5;
  const int rowPair = wave & 3;
  const int colHalf = wave >> 2;
  const int m0 = blockIdx.y * 128;
  const int n0 = blockIdx.x * 64;

  v8f a00 = {}, a01 = {}, a10 = {}, a11 = {};

  const half_t* Abase = A  + (size_t)m0 * K;
  const half_t* Bbase = Bt + (size_t)n0 * K;
  const int steps = K >> 5;

  stage_a_async(As[0], Abase, K, tid);
  stage_b_async(Bs[0], Bbase, K, tid);
  wait_async();
  __syncthreads();

  for (int s = 0; s < steps; ++s) {
    const int p = s & 1;
    if (s + 1 < steps) {
      stage_a_async(As[p ^ 1], Abase + (s + 1) * 32, K, tid);
      stage_b_async(Bs[p ^ 1], Bbase + (s + 1) * 32, K, tid);
    }
    wave_mma(As[p], Bs[p], rowPair, colHalf, lane, a00, a01, a10, a11);
    wait_async();
    __syncthreads();
  }

  const int mB = m0 + rowPair * 32 + (lane >> 4) * 8;
  const int n  = n0 + colHalf * 32 + (lane & 15);
  const float bi0 = bias[n], bi1 = bias[n + 16];
#pragma unroll
  for (int v = 0; v < 8; ++v) {
#pragma unroll
    for (int rs = 0; rs < 2; ++rs) {
      const int r = mB + v + rs * 16;
      float x0 = (rs ? a10[v] : a00[v]) + bi0;
      float x1 = (rs ? a11[v] : a01[v]) + bi1;
      if (mode == MODE_F32) {
        float* C = (float*)Cout;
        C[(size_t)r * N + n]      = x0;
        C[(size_t)r * N + n + 16] = x1;
      } else if (mode == MODE_VT) {
        // vt[((b*H + h)*64 + dv)*S + s]
        half_t* Vt = (half_t*)Cout;
        const int bb = r >> 8, ss = r & 255;
        const int h0 = n >> 6, d0 = n & 63;
        const int h1 = (n + 16) >> 6, d1 = (n + 16) & 63;
        Vt[((size_t)(bb * Hh + h0) * 64 + d0) * Ssz + ss] = (half_t)x0;
        Vt[((size_t)(bb * Hh + h1) * 64 + d1) * Ssz + ss] = (half_t)x1;
      } else {
        if (mode == MODE_F16_GELU) { x0 = gelu_f(x0); x1 = gelu_f(x1); }
        half_t* C = (half_t*)Cout;
        C[(size_t)r * N + n]      = (half_t)x0;
        C[(size_t)r * N + n + 16] = (half_t)x1;
      }
    }
  }
}

// =====================================================================
// Attention scores: scores[bh][q][k] = (Q.K^T)/8, masked, optional help
// mix. Q,K f16 in [B,S,H*DK]. Block tile 128(q) x 64(k), 2 K-steps.
// grid = (S/64, S/128, B*H), block = 256.
// =====================================================================
__global__ __launch_bounds__(256) void attn_scores_f16(
    const half_t* __restrict__ Q, const half_t* __restrict__ Km,
    const unsigned char* __restrict__ mask, const float* __restrict__ help,
    float* __restrict__ scores) {
  __shared__ __align__(16) half_t Qs[2][128 * LDF];
  __shared__ __align__(16) half_t Ks[2][64 * LDF];

  const int tid  = threadIdx.x;
  const int lane = tid & 31;
  const int wave = tid >> 5;
  const int rowPair = wave & 3;
  const int colHalf = wave >> 2;
  const int bh = blockIdx.z;
  const int b  = bh >> 3;
  const int h  = bh & 7;
  const int q0 = blockIdx.y * 128;
  const int k0 = blockIdx.x * 64;

  v8f a00 = {}, a01 = {}, a10 = {}, a11 = {};

  const half_t* qbase = Q  + ((size_t)(b * Ssz + q0) * Dsz + h * DKk);
  const half_t* kbase = Km + ((size_t)(b * Ssz + k0) * Dsz + h * DKk);

  stage_a_async(Qs[0], qbase, Dsz, tid);
  stage_b_async(Ks[0], kbase, Dsz, tid);
  wait_async();
  __syncthreads();
#pragma unroll
  for (int s = 0; s < 2; ++s) {
    const int p = s & 1;
    if (s + 1 < 2) {
      stage_a_async(Qs[p ^ 1], qbase + 32, Dsz, tid);
      stage_b_async(Ks[p ^ 1], kbase + 32, Dsz, tid);
    }
    wave_mma(Qs[p], Ks[p], rowPair, colHalf, lane, a00, a01, a10, a11);
    wait_async();
    __syncthreads();
  }

  const int qB = q0 + rowPair * 32 + (lane >> 4) * 8;
  const int kn = k0 + colHalf * 32 + (lane & 15);
#pragma unroll
  for (int v = 0; v < 8; ++v) {
#pragma unroll
    for (int rs = 0; rs < 2; ++rs) {
      const int qq = qB + v + rs * 16;
      float s0 = (rs ? a10[v] : a00[v]) * 0.125f;   // 1/sqrt(64)
      float s1 = (rs ? a11[v] : a01[v]) * 0.125f;
      const size_t mrow = ((size_t)b * Ssz + qq) * Ssz;
      if (mask[mrow + kn])      s0 = -1e9f;
      if (mask[mrow + kn + 16]) s1 = -1e9f;
      const size_t srow = ((size_t)bh * Ssz + qq) * Ssz;
      if (help) {
        float h0 = help[srow + kn], h1 = help[srow + kn + 16];
        float m0v = (h0 > -1e8f) ? 1.0f : 0.0f;
        float m1v = (h1 > -1e8f) ? 1.0f : 0.0f;
        s0 = h0 * (m0v * 0.5f) + s0 * (m0v * 0.5f + (1.0f - m0v));
        s1 = h1 * (m1v * 0.5f) + s1 * (m1v * 0.5f + (1.0f - m1v));
      }
      scores[srow + kn]      = s0;
      scores[srow + kn + 16] = s1;
    }
  }
}

// =====================================================================
// Row softmax over 256 elems, one row per wave32. If o16 != null, emit
// f16 probabilities (for the WMMA ctx matmul) instead of f32 in place.
// =====================================================================
__global__ __launch_bounds__(256) void softmax256(float* __restrict__ s,
                                                  half_t* __restrict__ o16) {
  const int row  = blockIdx.x * 8 + (threadIdx.x >> 5);
  const int lane = threadIdx.x & 31;
  float* p = s + (size_t)row * 256;
  float vals[8];
  float mx = -3.4e38f;
#pragma unroll
  for (int i = 0; i < 8; ++i) { vals[i] = p[lane + i * 32]; mx = fmaxf(mx, vals[i]); }
#pragma unroll
  for (int off = 16; off > 0; off >>= 1) mx = fmaxf(mx, __shfl_xor(mx, off, 32));
  float sum = 0.0f;
#pragma unroll
  for (int i = 0; i < 8; ++i) { vals[i] = __expf(vals[i] - mx); sum += vals[i]; }
#pragma unroll
  for (int off = 16; off > 0; off >>= 1) sum += __shfl_xor(sum, off, 32);
  const float inv = 1.0f / sum;
  if (o16) {
    half_t* q = o16 + (size_t)row * 256;
#pragma unroll
    for (int i = 0; i < 8; ++i) q[lane + i * 32] = (half_t)(vals[i] * inv);
  } else {
#pragma unroll
    for (int i = 0; i < 8; ++i) p[lane + i * 32] = vals[i] * inv;
  }
}

// =====================================================================
// ctx16[b][q][h*DV+d] = sum_k attn16[bh][q][k] * Vt[bh][d][k]
// Block tile 128(q) x 64(dv), 8 K-steps. grid = (S/128, B*H).
// =====================================================================
__global__ __launch_bounds__(256) void attn_ctx_f16(
    const half_t* __restrict__ attn, const half_t* __restrict__ Vt,
    half_t* __restrict__ ctx) {
  __shared__ __align__(16) half_t As[2][128 * LDF];
  __shared__ __align__(16) half_t Bs[2][64 * LDF];

  const int tid  = threadIdx.x;
  const int lane = tid & 31;
  const int wave = tid >> 5;
  const int rowPair = wave & 3;
  const int colHalf = wave >> 2;
  const int bh = blockIdx.y;
  const int b  = bh >> 3;
  const int h  = bh & 7;
  const int q0 = blockIdx.x * 128;

  v8f a00 = {}, a01 = {}, a10 = {}, a11 = {};

  const half_t* abase = attn + ((size_t)bh * Ssz + q0) * Ssz;
  const half_t* bbase = Vt + (size_t)bh * 64 * Ssz;

  stage_a_async(As[0], abase, Ssz, tid);
  stage_b_async(Bs[0], bbase, Ssz, tid);
  wait_async();
  __syncthreads();

  for (int s = 0; s < (Ssz >> 5); ++s) {
    const int p = s & 1;
    if (s + 1 < (Ssz >> 5)) {
      stage_a_async(As[p ^ 1], abase + (s + 1) * 32, Ssz, tid);
      stage_b_async(Bs[p ^ 1], bbase + (s + 1) * 32, Ssz, tid);
    }
    wave_mma(As[p], Bs[p], rowPair, colHalf, lane, a00, a01, a10, a11);
    wait_async();
    __syncthreads();
  }

  const int qB = q0 + rowPair * 32 + (lane >> 4) * 8;
  const int n  = colHalf * 32 + (lane & 15);
#pragma unroll
  for (int v = 0; v < 8; ++v) {
#pragma unroll
    for (int rs = 0; rs < 2; ++rs) {
      const int qq = qB + v + rs * 16;
      half_t* d = ctx + ((size_t)(b * Ssz + qq) * Dsz + h * DKk);
      d[n]      = (half_t)(rs ? a10[v] : a00[v]);
      d[n + 16] = (half_t)(rs ? a11[v] : a01[v]);
    }
  }
}

// =====================================================================
// out = LayerNorm(y + res)*g + beta (D=512), one row per wave32.
// Optional f16 copy for downstream WMMA consumers.
// =====================================================================
__global__ __launch_bounds__(256) void add_ln(
    const float* __restrict__ y, const float* __restrict__ res,
    const float* __restrict__ g, const float* __restrict__ beta,
    float* __restrict__ out, half_t* __restrict__ out16) {
  const int row  = blockIdx.x * 8 + (threadIdx.x >> 5);
  const int lane = threadIdx.x & 31;
  const float* py = y   + (size_t)row * Dsz;
  const float* pr = res + (size_t)row * Dsz;
  float v[16];
  float sum = 0.0f;
#pragma unroll
  for (int i = 0; i < 16; ++i) {
    v[i] = py[lane + i * 32] + pr[lane + i * 32];
    sum += v[i];
  }
#pragma unroll
  for (int off = 16; off > 0; off >>= 1) sum += __shfl_xor(sum, off, 32);
  const float mean = sum * (1.0f / Dsz);
  float var = 0.0f;
#pragma unroll
  for (int i = 0; i < 16; ++i) { float d = v[i] - mean; var += d * d; }
#pragma unroll
  for (int off = 16; off > 0; off >>= 1) var += __shfl_xor(var, off, 32);
  const float inv = rsqrtf(var * (1.0f / Dsz) + 1e-5f);
  float* po = out + (size_t)row * Dsz;
  half_t* ph = out16 ? out16 + (size_t)row * Dsz : (half_t*)0;
#pragma unroll
  for (int i = 0; i < 16; ++i) {
    const int c = lane + i * 32;
    const float r = (v[i] - mean) * inv * g[c] + beta[c];
    po[c] = r;
    if (ph) ph[c] = (half_t)r;
  }
}

// f32 -> f16 flat convert, 8 elems/thread (exact grid)
__global__ __launch_bounds__(256) void cvt16(const float* __restrict__ x,
                                             half_t* __restrict__ y) {
  const size_t i = ((size_t)blockIdx.x * 256 + threadIdx.x) * 8;
  float4 a = ((const float4*)(x + i))[0];
  float4 b = ((const float4*)(x + i))[1];
  half_t* d = y + i;
  d[0] = (half_t)a.x; d[1] = (half_t)a.y; d[2] = (half_t)a.z; d[3] = (half_t)a.w;
  d[4] = (half_t)b.x; d[5] = (half_t)b.y; d[6] = (half_t)b.z; d[7] = (half_t)b.w;
}

// W[K][N] f32 -> Wt[N][K] f16, 32x32 LDS tile transpose. grid=(N/32,K/32)
__global__ __launch_bounds__(256) void wtrans(const float* __restrict__ W,
                                              half_t* __restrict__ Wt,
                                              int K, int N) {
  __shared__ half_t t[32][36];
  const int tid = threadIdx.x;
  const int k0 = blockIdx.y << 5, n0 = blockIdx.x << 5;
  const int r = tid >> 3, c = (tid & 7) << 2;
  float4 f = *(const float4*)(W + (size_t)(k0 + r) * N + n0 + c);
  t[c + 0][r] = (half_t)f.x;
  t[c + 1][r] = (half_t)f.y;
  t[c + 2][r] = (half_t)f.z;
  t[c + 3][r] = (half_t)f.w;
  __syncthreads();
  half_t* d = Wt + (size_t)(n0 + r) * K + k0 + c;
  d[0] = t[r][c]; d[1] = t[r][c + 1]; d[2] = t[r][c + 2]; d[3] = t[r][c + 3];
}

// =====================================================================
// Host-side orchestration
// =====================================================================
// Param tensor index inside d_in: 1 + layer*32 + branch*16 + slot
// slots: 0 wq, 1 bq, 2 wk, 3 bk, 4 wv, 5 bv, 6 wo, 7 bo,
//        8 ln1_g, 9 ln1_b, 10 w1, 11 b1, 12 w2, 13 b2, 14 ln2_g, 15 ln2_b
static inline const float* PP(void* const* d_in, int layer, int branch, int slot) {
  return (const float*)d_in[1 + layer * 32 + branch * 16 + slot];
}

extern "C" void kernel_launch(void* const* d_in, const int* in_sizes, int n_in,
                              void* d_out, int out_size, void* d_ws, size_t ws_size,
                              hipStream_t stream) {
  (void)in_sizes; (void)n_in; (void)out_size; (void)ws_size;

  const float* x = (const float*)d_in[0];
  const unsigned char* seq_mask = (const unsigned char*)d_in[1 + Ll * 32];
  const unsigned char* dep_mask = (const unsigned char*)d_in[2 + Ll * 32];

  // ---- workspace carve-up ----
  char* w = (char*)d_ws;
  const size_t MDf  = (size_t)Mrows * Dsz * sizeof(float);            // 32 MB
  const size_t MDh  = (size_t)Mrows * Dsz * sizeof(half_t);           // 16 MB
  const size_t ATTf = (size_t)Bsz * Hh * Ssz * Ssz * sizeof(float);   // 128 MB
  const size_t ATTh = (size_t)Bsz * Hh * Ssz * Ssz * sizeof(half_t);  //  64 MB
  half_t* q16    = (half_t*)w; w += MDh;
  half_t* k16    = (half_t*)w; w += MDh;
  half_t* vt     = (half_t*)w; w += MDh;   // [bh][dv][s]
  float*  depS   = (float*)w;  w += ATTf;  // dep scores -> softmax -> help
  float*  seqS   = (float*)w;  w += ATTf;
  half_t* attn16 = (half_t*)w; w += ATTh;
  half_t* ctx16  = (half_t*)w; w += MDh;
  float*  ob     = (float*)w;  w += MDf;
  float*  aob    = (float*)w;  w += MDf;
  half_t* aob16  = (half_t*)w; w += MDh;
  half_t* ffh16  = (half_t*)w; w += (size_t)Mrows * FFs * sizeof(half_t);
  float*  f2b    = (float*)w;  w += MDf;
  float*  hb     = (float*)w;  w += MDf;
  half_t* h16    = (half_t*)w; w += MDh;
  // f16 transposed weights: per layer {dep wq, dep wk, seq wq, wk, wv, wo, w1, w2}
  half_t* wt[Ll][8];
  for (int l = 0; l < Ll; ++l)
    for (int i = 0; i < 8; ++i) {
      size_t elems = (i < 6) ? (size_t)Dsz * Dsz : (size_t)Dsz * FFs;
      wt[l][i] = (half_t*)w; w += elems * sizeof(half_t);
    }

  const dim3 blk(256);
  const dim3 gD(Dsz / 64, Mrows / 128);
  const dim3 gF1(FFs / 64, Mrows / 128);
  const dim3 gSc(Ssz / 64, Ssz / 128, Bsz * Hh);
  const dim3 gCx(Ssz / 128, Bsz * Hh);
  const int smBlocks = (Bsz * Hh * Ssz) / 8;
  const int lnBlocks = Mrows / 8;
  const dim3 gT512(Dsz / 32, Dsz / 32);     // 512x512 transpose
  const dim3 gTw1(FFs / 32, Dsz / 32);      // K=512, N=2048
  const dim3 gTw2(Dsz / 32, FFs / 32);      // K=2048, N=512

  // ---- one-time prep: f16 transposed weights + f16 input ----
  const int wslot[8] = {0, 2, 0, 2, 4, 6, 10, 12};   // param slot per wt index
  for (int l = 0; l < Ll; ++l) {
    hipLaunchKernelGGL(wtrans, gT512, blk, 0, stream, PP(d_in, l, 0, 0), wt[l][0], Dsz, Dsz);
    hipLaunchKernelGGL(wtrans, gT512, blk, 0, stream, PP(d_in, l, 0, 2), wt[l][1], Dsz, Dsz);
    hipLaunchKernelGGL(wtrans, gT512, blk, 0, stream, PP(d_in, l, 1, 0), wt[l][2], Dsz, Dsz);
    hipLaunchKernelGGL(wtrans, gT512, blk, 0, stream, PP(d_in, l, 1, 2), wt[l][3], Dsz, Dsz);
    hipLaunchKernelGGL(wtrans, gT512, blk, 0, stream, PP(d_in, l, 1, 4), wt[l][4], Dsz, Dsz);
    hipLaunchKernelGGL(wtrans, gT512, blk, 0, stream, PP(d_in, l, 1, 6), wt[l][5], Dsz, Dsz);
    hipLaunchKernelGGL(wtrans, gTw1, blk, 0, stream, PP(d_in, l, 1, 10), wt[l][6], Dsz, FFs);
    hipLaunchKernelGGL(wtrans, gTw2, blk, 0, stream, PP(d_in, l, 1, 12), wt[l][7], FFs, Dsz);
  }
  (void)wslot;
  hipLaunchKernelGGL(cvt16, dim3((Mrows * Dsz) / (256 * 8)), blk, 0, stream, x, h16);

  const float* hres = x;
  for (int l = 0; l < Ll; ++l) {
    // ---- dep branch: only attn probabilities are live downstream ----
    hipLaunchKernelGGL(gemm_f16, gD, blk, 0, stream,
        h16, wt[l][0], PP(d_in, l, 0, 1), (void*)q16, Mrows, Dsz, Dsz, MODE_F16);
    hipLaunchKernelGGL(gemm_f16, gD, blk, 0, stream,
        h16, wt[l][1], PP(d_in, l, 0, 3), (void*)k16, Mrows, Dsz, Dsz, MODE_F16);
    hipLaunchKernelGGL(attn_scores_f16, gSc, blk, 0, stream,
        q16, k16, dep_mask, (const float*)nullptr, depS);
    hipLaunchKernelGGL(softmax256, dim3(smBlocks), blk, 0, stream, depS, (half_t*)nullptr);

    // ---- seq branch ----
    hipLaunchKernelGGL(gemm_f16, gD, blk, 0, stream,
        h16, wt[l][2], PP(d_in, l, 1, 1), (void*)q16, Mrows, Dsz, Dsz, MODE_F16);
    hipLaunchKernelGGL(gemm_f16, gD, blk, 0, stream,
        h16, wt[l][3], PP(d_in, l, 1, 3), (void*)k16, Mrows, Dsz, Dsz, MODE_F16);
    hipLaunchKernelGGL(gemm_f16, gD, blk, 0, stream,
        h16, wt[l][4], PP(d_in, l, 1, 5), (void*)vt, Mrows, Dsz, Dsz, MODE_VT);
    hipLaunchKernelGGL(attn_scores_f16, gSc, blk, 0, stream,
        q16, k16, seq_mask, depS, seqS);
    hipLaunchKernelGGL(softmax256, dim3(smBlocks), blk, 0, stream, seqS, attn16);
    hipLaunchKernelGGL(attn_ctx_f16, gCx, blk, 0, stream, attn16, vt, ctx16);
    hipLaunchKernelGGL(gemm_f16, gD, blk, 0, stream,
        ctx16, wt[l][5], PP(d_in, l, 1, 7), (void*)ob, Mrows, Dsz, Dsz, MODE_F32);
    hipLaunchKernelGGL(add_ln, dim3(lnBlocks), blk, 0, stream,
        ob, hres, PP(d_in, l, 1, 8), PP(d_in, l, 1, 9), aob, aob16);

    // ---- seq FFN ----
    hipLaunchKernelGGL(gemm_f16, gF1, blk, 0, stream,
        aob16, wt[l][6], PP(d_in, l, 1, 11), (void*)ffh16, Mrows, FFs, Dsz, MODE_F16_GELU);
    hipLaunchKernelGGL(gemm_f16, gD, blk, 0, stream,
        ffh16, wt[l][7], PP(d_in, l, 1, 13), (void*)f2b, Mrows, Dsz, FFs, MODE_F32);
    float*  h_out  = (l == Ll - 1) ? (float*)d_out : hb;
    half_t* h_out16 = (l == Ll - 1) ? (half_t*)nullptr : h16;
    hipLaunchKernelGGL(add_ln, dim3(lnBlocks), blk, 0, stream,
        f2b, aob, PP(d_in, l, 1, 14), PP(d_in, l, 1, 15), h_out, h_out16);
    hres = h_out;
  }
}
